// selfAttention_63367947485365
// MI455X (gfx1250) — compile-verified, matrix operand order checked by reading, not run on hardware
//
#include <hip/hip_runtime.h>
#include <hip/hip_bf16.h>
#include <math.h>
#include <stdint.h>

// Problem constants (reference: B=4, S=1024, D=1024, H=16, HD=64)
#define B_   4
#define S_   1024
#define D_   1024
#define H_   16
#define HD_  64
#define NTOK (B_ * S_)   // 4096 tokens

typedef unsigned short u16;
typedef __attribute__((ext_vector_type(8)))  u16    u16x8;
typedef __attribute__((ext_vector_type(16))) __bf16 v16bf;
typedef __attribute__((ext_vector_type(8)))  float  v8f;

// ---- helpers ---------------------------------------------------------------

__device__ __forceinline__ u16 f2bf(float f) {
  union { float f; unsigned u; } c; c.f = f;
  unsigned u = c.u;
  u += 0x7FFFu + ((u >> 16) & 1u);   // round-to-nearest-even
  return (u16)(u >> 16);
}

// Build a 16x32 (or 32x16) bf16 WMMA fragment from two contiguous 16B chunks.
// Per ISA layout (16-bit A 16x32): lane-lo holds K=0..7 (v0-3) and K=16..23
// (v4-7); lane-hi holds K=8..15 and K=24..31.
__device__ __forceinline__ v16bf make_frag(const u16* p0, const u16* p1) {
  union { u16x8 h[2]; v16bf v; } u;
  u.h[0] = *(const u16x8*)p0;
  u.h[1] = *(const u16x8*)p1;
  return u.v;
}

__device__ __forceinline__ v8f wmma_bf16(v16bf a, v16bf b, v8f c) {
  return __builtin_amdgcn_wmma_f32_16x16x32_bf16(
      false, a, false, b, (short)0, c, false, false);
}

// Async copy of one 16B chunk from global to LDS (ASYNCcnt-tracked).
__device__ __forceinline__ void async_cp16(unsigned lds_off, const u16* g) {
  asm volatile("global_load_async_to_lds_b128 %0, %1, off"
               :: "v"(lds_off), "v"(g) : "memory");
}
__device__ __forceinline__ void wait_async0() {
  asm volatile("s_wait_asynccnt 0x0" ::: "memory");
}
__device__ __forceinline__ unsigned lds_off32(const void* p) {
  return (unsigned)(uintptr_t)p;   // low 32 bits of generic ptr = LDS offset
}

// ---- conversion kernels ----------------------------------------------------

__global__ void cvt_act(const float* __restrict__ x, u16* __restrict__ y, int n) {
  int i = blockIdx.x * blockDim.x + threadIdx.x;
  if (i < n) y[i] = f2bf(x[i]);
}

// W is [K][N] row-major f32; produce WT[N][K] bf16 so a B-fragment column
// reads contiguous K.
__global__ void cvt_wT(const float* __restrict__ w, u16* __restrict__ wt) {
  int i = blockIdx.x * blockDim.x + threadIdx.x;
  int n = i & (D_ - 1);
  int k = i >> 10;
  wt[(size_t)n * D_ + k] = f2bf(w[(size_t)k * D_ + n]);
}

// ---- fused QKV projection GEMM --------------------------------------------
// Each wave: 16x64 tile.  All 8 waves in a WG share the same 16-row A block,
// async-staged into LDS once (32 KB).  K-loop is unrolled x2 with disjoint
// fragment register sets -> true double buffering, no rotation movs.

__global__ void qkv_gemm(const u16* __restrict__ eegb, const u16* __restrict__ picb,
                         const u16* __restrict__ WkT,  const u16* __restrict__ WqT,
                         const u16* __restrict__ WvT,
                         const float* __restrict__ bk, const float* __restrict__ bq,
                         const float* __restrict__ bv,
                         u16* __restrict__ qo, u16* __restrict__ ko,
                         u16* __restrict__ vTo) {
  __shared__ u16 ldsA[16 * D_];          // 32 KB: 16 rows x 1024 cols bf16
  const int mode = blockIdx.y;           // 0=q 1=k 2=v
  const int lane = threadIdx.x & 31;
  const int wv   = threadIdx.x >> 5;
  const int l16  = lane & 15;
  const int hig  = lane >> 4;

  const int tile = blockIdx.x * 8 + wv;
  const int row0 = (tile >> 4) << 4;     // shared by all 8 waves in this WG
  const int col0 = (tile & 15) << 6;

  const u16*  A    = (mode == 1) ? eegb : picb;
  const u16*  WT   = (mode == 0) ? WqT : (mode == 1 ? WkT : WvT);
  const float* bias = (mode == 0) ? bq : (mode == 1 ? bk : bv);

  // ---- async-stage A row block (contiguous 16*1024 bf16 = 32 KB) ----
  {
    const u16* gbase = A + (size_t)row0 * D_;
    const unsigned lbase = lds_off32(&ldsA[0]);
    const int tid = threadIdx.x;
#pragma unroll
    for (int j = 0; j < 8; ++j) {
      const int c = tid * 8 + j;               // 16B chunk 0..2047
      async_cp16(lbase + c * 16, gbase + c * 8);
    }
    wait_async0();
    __syncthreads();
  }

  v8f acc[4] = {};
  const u16* arow = &ldsA[l16 * D_];
  const u16* wcol[4];
#pragma unroll
  for (int c = 0; c < 4; ++c)
    wcol[c] = WT + (size_t)(col0 + c * 16 + l16) * D_ + hig * 8;

  auto ldA = [&](int kk) {
    return make_frag(arow + kk + hig * 8, arow + kk + 16 + hig * 8);
  };
  auto ldB = [&](v16bf* dst, int kk) {
#pragma unroll
    for (int c = 0; c < 4; ++c)
      dst[c] = make_frag(wcol[c] + kk, wcol[c] + kk + 16);
  };

  // two-phase double-buffered pipeline (no register rotation)
  v16bf af0 = ldA(0), af1;
  v16bf bf0[4], bf1[4];
  ldB(bf0, 0);

  for (int kk = 0; kk < D_; kk += 64) {
    af1 = ldA(kk + 32);
    ldB(bf1, kk + 32);
    if (kk + 96 < D_) __builtin_prefetch(wcol[0] + kk + 96, 0, 1);
#pragma unroll
    for (int c = 0; c < 4; ++c) acc[c] = wmma_bf16(af0, bf0[c], acc[c]);
    if (kk + 64 < D_) {
      af0 = ldA(kk + 64);
      ldB(bf0, kk + 64);
    }
#pragma unroll
    for (int c = 0; c < 4; ++c) acc[c] = wmma_bf16(af1, bf1[c], acc[c]);
  }

#pragma unroll
  for (int c = 0; c < 4; ++c) {
    const int n = col0 + c * 16 + l16;
    const float bval = bias[n];
    const int h = n >> 6, d = n & 63;
#pragma unroll
    for (int r = 0; r < 8; ++r) {
      const int rowg = row0 + r + hig * 8;
      const int b = rowg >> 10, s = rowg & (S_ - 1);
      const float val = acc[c][r] + bval;
      if (mode == 2) {
        vTo[(((size_t)(b * H_ + h) * HD_ + d) * S_) + s] = f2bf(val);   // [b,h,hd,S]
      } else {
        u16* dst = (mode == 0) ? qo : ko;                               // [b,h,S,hd]
        dst[(((size_t)(b * H_ + h) * S_ + s) * HD_) + d] = f2bf(val);
      }
    }
  }
}

// ---- flash attention -------------------------------------------------------
// grid.x = B*H*(S/128) = 512; block = 256 (8 waves); each wave: 16 q-rows.
// K/V tiles for each 32-key block are shared by all 8 waves -> double-buffered
// async staging into LDS: issue tile i+1, compute on tile i, 1 barrier/iter.

__global__ void flash_attn(const u16* __restrict__ q, const u16* __restrict__ k,
                           const u16* __restrict__ vT, u16* __restrict__ ctx) {
  __shared__ u16 kt[2][32 * 64];    // 2 x 4 KB  [key][d]
  __shared__ u16 vt[2][64 * 32];    // 2 x 4 KB  [d][key]
  __shared__ u16 plds[8 * 512];     // 8 KB per-wave 16x32 P tiles

  const int tid  = threadIdx.x;
  const int lane = tid & 31;
  const int wv   = tid >> 5;
  const int l16  = lane & 15;
  const int hig  = lane >> 4;

  const int qblk = blockIdx.x & 7;
  const int bh   = blockIdx.x >> 3;
  const int h    = bh & (H_ - 1);
  const int b0   = bh >> 4;
  const int qbase = qblk * 128 + wv * 16;

  const u16* qh = q  + (size_t)bh * S_ * HD_;
  const u16* kh = k  + (size_t)bh * S_ * HD_;
  const u16* vh = vT + (size_t)bh * HD_ * S_;

  const u16* qp = qh + (size_t)(qbase + l16) * HD_;
  v16bf qa0 = make_frag(qp +      hig * 8, qp + 16 + hig * 8);  // d 0..31
  v16bf qa1 = make_frag(qp + 32 + hig * 8, qp + 48 + hig * 8);  // d 32..63

  float m[8], lsum[8];
#pragma unroll
  for (int r = 0; r < 8; ++r) { m[r] = -3.0e38f; lsum[r] = 0.f; }
  v8f o[4] = {};

  u16* pw = plds + wv * 512;
  const float SC = 0.125f;                   // 1/sqrt(64)

  // cooperative async stage of one 32-key K/V tile pair (8 KB = 2 chunks/thr)
  auto issue_tiles = [&](int n0, int bsel) {
    // K tile: 32 keys x 64 d, contiguous 4 KB at kh + n0*64
    async_cp16(lds_off32(&kt[bsel][0]) + tid * 16,
               kh + (size_t)n0 * HD_ + tid * 8);
    // V tile: 64 d-rows x 32 keys (64B per row, stride S_)
    const int d = tid >> 2, part = tid & 3;
    async_cp16(lds_off32(&vt[bsel][0]) + (d * 32 + part * 8) * 2,
               vh + (size_t)d * S_ + n0 + part * 8);
  };

  issue_tiles(0, 0);

  for (int it = 0; it < S_ / 32; ++it) {
    const int n0   = it * 32;
    const int bsel = it & 1;
    wait_async0();         // own chunks of tile `it` landed
    __syncthreads();       // everyone's chunks landed; prev compute done
    if (it + 1 < S_ / 32) issue_tiles(n0 + 32, bsel ^ 1);

    // ---- scores S = Q @ K^T for 16x32 tile (K from LDS) ----
    const u16* kp0 = &kt[bsel][(size_t)l16 * 64];
    const u16* kp1 = &kt[bsel][(size_t)(16 + l16) * 64];
    v16bf kb00 = make_frag(kp0 +      hig * 8, kp0 + 16 + hig * 8);
    v16bf kb01 = make_frag(kp0 + 32 + hig * 8, kp0 + 48 + hig * 8);
    v16bf kb10 = make_frag(kp1 +      hig * 8, kp1 + 16 + hig * 8);
    v16bf kb11 = make_frag(kp1 + 32 + hig * 8, kp1 + 48 + hig * 8);

    v8f c0 = {}, c1 = {};
    c0 = wmma_bf16(qa0, kb00, c0);
    c0 = wmma_bf16(qa1, kb01, c0);
    c1 = wmma_bf16(qa0, kb10, c1);
    c1 = wmma_bf16(qa1, kb11, c1);

    // ---- online softmax (row stats via 16-lane shuffle reductions) ----
#pragma unroll
    for (int r = 0; r < 8; ++r) {
      float s0 = c0[r] * SC, s1 = c1[r] * SC;
      float t = fmaxf(s0, s1);
      t = fmaxf(t, __shfl_xor(t, 1, 32));
      t = fmaxf(t, __shfl_xor(t, 2, 32));
      t = fmaxf(t, __shfl_xor(t, 4, 32));
      t = fmaxf(t, __shfl_xor(t, 8, 32));
      float mnew = fmaxf(m[r], t);
      float scl  = __expf(m[r] - mnew);
      m[r] = mnew;
      float p0 = __expf(s0 - mnew);
      float p1 = __expf(s1 - mnew);
      float ps = p0 + p1;
      ps += __shfl_xor(ps, 1, 32);
      ps += __shfl_xor(ps, 2, 32);
      ps += __shfl_xor(ps, 4, 32);
      ps += __shfl_xor(ps, 8, 32);
      lsum[r] = lsum[r] * scl + ps;
      o[0][r] *= scl; o[1][r] *= scl; o[2][r] *= scl; o[3][r] *= scl;
      // stage P (C-layout -> LDS row-major [row][key32]) as bf16
      const int rr = r + hig * 8;
      pw[rr * 32 + l16]      = f2bf(p0);
      pw[rr * 32 + 16 + l16] = f2bf(p1);
    }
    asm volatile("s_wait_dscnt 0x0" ::: "memory");

    // re-read P in A-fragment layout (wave-private LDS region)
    const u16* pr = pw + l16 * 32;
    v16bf pa = make_frag(pr + hig * 8, pr + 16 + hig * 8);

    // ---- O += P @ V  (V from LDS, [d][key] tile) ----
#pragma unroll
    for (int j = 0; j < 4; ++j) {
      const u16* vp = &vt[bsel][(size_t)(j * 16 + l16) * 32];
      v16bf vb = make_frag(vp + hig * 8, vp + 16 + hig * 8);
      o[j] = wmma_bf16(pa, vb, o[j]);
    }
  }

  // ---- normalize & store ctx as bf16 in [B,S,D] ----
#pragma unroll
  for (int r = 0; r < 8; ++r) {
    const float inv = 1.0f / lsum[r];
    const int rowg = qbase + r + hig * 8;
    u16* crow = ctx + ((size_t)(b0 * S_ + rowg)) * D_ + h * HD_;
    crow[0 * 16 + l16] = f2bf(o[0][r] * inv);
    crow[1 * 16 + l16] = f2bf(o[1][r] * inv);
    crow[2 * 16 + l16] = f2bf(o[2][r] * inv);
    crow[3 * 16 + l16] = f2bf(o[3][r] * inv);
  }
}

// ---- output projection + residual -----------------------------------------

__global__ void out_gemm(const u16* __restrict__ ctxb, const u16* __restrict__ WdT,
                         const float* __restrict__ bd, const float* __restrict__ pic,
                         float* __restrict__ xo) {
  __shared__ u16 ldsA[16 * D_];          // 32 KB
  const int lane = threadIdx.x & 31;
  const int wv   = threadIdx.x >> 5;
  const int l16  = lane & 15;
  const int hig  = lane >> 4;

  const int tile = blockIdx.x * 8 + wv;
  const int row0 = (tile >> 4) << 4;
  const int col0 = (tile & 15) << 6;

  {
    const u16* gbase = ctxb + (size_t)row0 * D_;
    const unsigned lbase = lds_off32(&ldsA[0]);
    const int tid = threadIdx.x;
#pragma unroll
    for (int j = 0; j < 8; ++j) {
      const int c = tid * 8 + j;
      async_cp16(lbase + c * 16, gbase + c * 8);
    }
    wait_async0();
    __syncthreads();
  }

  v8f acc[4] = {};
  const u16* arow = &ldsA[l16 * D_];
  const u16* wcol[4];
#pragma unroll
  for (int c = 0; c < 4; ++c)
    wcol[c] = WdT + (size_t)(col0 + c * 16 + l16) * D_ + hig * 8;

  auto ldA = [&](int kk) {
    return make_frag(arow + kk + hig * 8, arow + kk + 16 + hig * 8);
  };
  auto ldB = [&](v16bf* dst, int kk) {
#pragma unroll
    for (int c = 0; c < 4; ++c)
      dst[c] = make_frag(wcol[c] + kk, wcol[c] + kk + 16);
  };

  v16bf af0 = ldA(0), af1;
  v16bf bf0[4], bf1[4];
  ldB(bf0, 0);

  for (int kk = 0; kk < D_; kk += 64) {
    af1 = ldA(kk + 32);
    ldB(bf1, kk + 32);
    if (kk + 96 < D_) __builtin_prefetch(wcol[0] + kk + 96, 0, 1);
#pragma unroll
    for (int c = 0; c < 4; ++c) acc[c] = wmma_bf16(af0, bf0[c], acc[c]);
    if (kk + 64 < D_) {
      af0 = ldA(kk + 64);
      ldB(bf0, kk + 64);
    }
#pragma unroll
    for (int c = 0; c < 4; ++c) acc[c] = wmma_bf16(af1, bf1[c], acc[c]);
  }

#pragma unroll
  for (int c = 0; c < 4; ++c) {
    const int n = col0 + c * 16 + l16;
    const float bval = bd[n];
#pragma unroll
    for (int r = 0; r < 8; ++r) {
      const int rowg = row0 + r + hig * 8;
      const size_t idx = (size_t)rowg * D_ + n;
      xo[idx] = acc[c][r] + bval + pic[idx];     // residual
    }
  }
}

// ---- layer norm ------------------------------------------------------------

__global__ void ln_kernel(const float* __restrict__ x, const float* __restrict__ w,
                          const float* __restrict__ bw, float* __restrict__ out) {
  __shared__ float red[256];
  const int row = blockIdx.x;
  const float* xr = x + (size_t)row * D_;

  float s = 0.f;
  for (int i = threadIdx.x; i < D_; i += 256) s += xr[i];
  red[threadIdx.x] = s; __syncthreads();
  for (int off = 128; off > 0; off >>= 1) {
    if (threadIdx.x < off) red[threadIdx.x] += red[threadIdx.x + off];
    __syncthreads();
  }
  const float mean = red[0] * (1.0f / D_);
  __syncthreads();

  float v = 0.f;
  for (int i = threadIdx.x; i < D_; i += 256) {
    float d = xr[i] - mean; v += d * d;
  }
  red[threadIdx.x] = v; __syncthreads();
  for (int off = 128; off > 0; off >>= 1) {
    if (threadIdx.x < off) red[threadIdx.x] += red[threadIdx.x + off];
    __syncthreads();
  }
  const float inv = rsqrtf(red[0] * (1.0f / D_) + 1e-12f);

  for (int i = threadIdx.x; i < D_; i += 256)
    out[(size_t)row * D_ + i] = w[i] * ((xr[i] - mean) * inv) + bw[i];
}

// ---- launch ---------------------------------------------------------------

extern "C" void kernel_launch(void* const* d_in, const int* in_sizes, int n_in,
                              void* d_out, int out_size, void* d_ws, size_t ws_size,
                              hipStream_t stream) {
  const float* eeg = (const float*)d_in[0];
  const float* pic = (const float*)d_in[1];
  const float* Wk  = (const float*)d_in[2];
  const float* bk  = (const float*)d_in[3];
  const float* Wq  = (const float*)d_in[4];
  const float* bq  = (const float*)d_in[5];
  const float* Wv  = (const float*)d_in[6];
  const float* bv  = (const float*)d_in[7];
  const float* Wd  = (const float*)d_in[8];
  const float* bd  = (const float*)d_in[9];
  const float* lnw = (const float*)d_in[10];
  const float* lnb = (const float*)d_in[11];

  char* ws = (char*)d_ws;
  u16*  eegb = (u16*)(ws + ((size_t)0  << 20));   // 8 MB
  u16*  picb = (u16*)(ws + ((size_t)8  << 20));   // 8 MB
  u16*  WkT  = (u16*)(ws + ((size_t)16 << 20));   // 2 MB
  u16*  WqT  = (u16*)(ws + ((size_t)18 << 20));   // 2 MB
  u16*  WvT  = (u16*)(ws + ((size_t)20 << 20));   // 2 MB
  u16*  WdT  = (u16*)(ws + ((size_t)22 << 20));   // 2 MB
  u16*  qb   = (u16*)(ws + ((size_t)24 << 20));   // 8 MB [b,h,S,hd]
  u16*  kbuf = (u16*)(ws + ((size_t)32 << 20));   // 8 MB [b,h,S,hd]
  u16*  vTb  = (u16*)(ws + ((size_t)40 << 20));   // 8 MB [b,h,hd,S]
  u16*  ctxb = (u16*)(ws + ((size_t)48 << 20));   // 8 MB [B,S,D]
  float* xb  = (float*)(ws + ((size_t)56 << 20)); // 16 MB

  const int nact = NTOK * D_;
  cvt_act<<<nact / 256, 256, 0, stream>>>(eeg, eegb, nact);
  cvt_act<<<nact / 256, 256, 0, stream>>>(pic, picb, nact);
  cvt_wT<<<(D_ * D_) / 256, 256, 0, stream>>>(Wk, WkT);
  cvt_wT<<<(D_ * D_) / 256, 256, 0, stream>>>(Wq, WqT);
  cvt_wT<<<(D_ * D_) / 256, 256, 0, stream>>>(Wv, WvT);
  cvt_wT<<<(D_ * D_) / 256, 256, 0, stream>>>(Wd, WdT);

  qkv_gemm<<<dim3(512, 3), 256, 0, stream>>>(eegb, picb, WkT, WqT, WvT,
                                             bk, bq, bv, qb, kbuf, vTb);
  flash_attn<<<512, 256, 0, stream>>>(qb, kbuf, vTb, ctxb);
  out_gemm<<<512, 256, 0, stream>>>(ctxb, WdT, bd, pic, xb);
  ln_kernel<<<NTOK, 256, 0, stream>>>(xb, lnw, lnb, (float*)d_out);
}
